// TransformerDecoderBlock_31138512896478
// MI455X (gfx1250) — matrix operand order repaired and checked by reading, compiled, not run on hardware
//
#include <hip/hip_runtime.h>
#include <hip/hip_bf16.h>
#include <stdint.h>

// ---------------------------------------------------------------------------
// Types for CDNA5 WMMA (wave32): D(16x16 f32) = A(16x32 bf16) * B(32x16 bf16) + C
// ---------------------------------------------------------------------------
typedef __attribute__((ext_vector_type(16))) __bf16 v16bf;
typedef __attribute__((ext_vector_type(8)))  float  v8f;
typedef __attribute__((ext_vector_type(4)))  uint32_t v4u;
typedef __attribute__((ext_vector_type(8)))  int32_t  v8i;
typedef __attribute__((ext_vector_type(4)))  int32_t  v4i;

union FragBF { v16bf v; uint32_t u[8]; uint16_t h[16]; };
union AccF  { v8f   v; float    f[8]; };

#if defined(__has_builtin)
#if __has_builtin(__builtin_amdgcn_tensor_load_to_lds) && __has_builtin(__builtin_amdgcn_s_wait_tensorcnt)
#define USE_TDM 1
#endif
#endif
#ifndef USE_TDM
#define USE_TDM 0
#endif

static __device__ __forceinline__ uint16_t to_bf16(float x) {
  uint32_t u = __float_as_uint(x);
  uint32_t r = u + 0x7FFFu + ((u >> 16) & 1u);  // round-to-nearest-even
  return (uint16_t)(r >> 16);
}

// A-operand fragment (16 rows x 32 K, 16-bit). Per ISA: lane r holds M=r,
// half 0 -> K [h*8 .. h*8+7] and [16+h*8 .. 16+h*8+7]. Two ds_load_b128.
static __device__ __forceinline__ void load_frag_a(const uint16_t* base, int stride, FragBF& f) {
  const int lane = threadIdx.x & 31;
  const int r = lane & 15, hf = lane >> 4;
  const uint16_t* p = base + r * stride + hf * 8;
  uint4 lo = *reinterpret_cast<const uint4*>(p);
  uint4 hi = *reinterpret_cast<const uint4*>(p + 16);
  f.u[0] = lo.x; f.u[1] = lo.y; f.u[2] = lo.z; f.u[3] = lo.w;
  f.u[4] = hi.x; f.u[5] = hi.y; f.u[6] = hi.z; f.u[7] = hi.w;
}

// B-operand fragment (N rows x 32 K in memory; logical 32x16). Lane r holds
// N=r; half h -> K [h*8 .. h*8+7] and [16+h*8 .. 16+h*8+7]. Two ds_load_b128.
static __device__ __forceinline__ void load_frag_b(const uint16_t* base, int stride, FragBF& f) {
  const int lane = threadIdx.x & 31;
  const int r = lane & 15, hf = lane >> 4;
  const uint16_t* p = base + r * stride + hf * 8;
  uint4 lo = *reinterpret_cast<const uint4*>(p);
  uint4 hi = *reinterpret_cast<const uint4*>(p + 16);
  f.u[0] = lo.x; f.u[1] = lo.y; f.u[2] = lo.z; f.u[3] = lo.w;
  f.u[4] = hi.x; f.u[5] = hi.y; f.u[6] = hi.z; f.u[7] = hi.w;
}

static __device__ __forceinline__ v8f wmma_bf16(const FragBF& a, const FragBF& b, v8f c) {
  return __builtin_amdgcn_wmma_f32_16x16x32_bf16(false, a.v, false, b.v, (short)0, c, false, false);
}

#if USE_TDM
// Issue one TDM 2D tile load: rows x 32 bf16 elements, row stride = Kdim elems,
// into LDS at lds_off with +16B padding after every 64B row (row pitch 80B = 40 elems).
// D# packing per CDNA5 ISA ch.8 (group0 128b, group1 256b, groups2/3 unused -> 0).
// This toolchain's builtin takes 6 args: (v4u g0, v8i g1, v4i g2, v4i g3, v8i gx, i32 cpol).
static __device__ __forceinline__ void tdm_load_tile_128x32(uint32_t lds_off,
                                                            const uint16_t* gptr,
                                                            uint32_t Kdim,
                                                            uint32_t rows_total) {
  const uint64_t ga = (uint64_t)(uintptr_t)gptr;
  v4u g0;
  g0.x = 1u;                                   // count=1, user descriptor
  g0.y = lds_off;                              // lds_addr (bytes)
  g0.z = (uint32_t)ga;                         // global_addr[31:0]
  g0.w = (uint32_t)((ga >> 32) & 0x1FFFFFFu)   // global_addr[56:32]
         | 0x80000000u;                        // type=2 ("image") in bits 127:126
  v8i g1;
  g1[0] = (int)((1u << 16)                     // data_size = 1 -> 2 bytes
              | (1u << 20)                     // pad_enable
              | (3u << 22)                     // pad_interval: 16 DWORDs (64B)
              | (3u << 25));                   // pad_amount: 4 DWORDs (16B)
  g1[1] = (int)((Kdim & 0xFFFFu) << 16);                             // tensor_dim0 lo16 (bits 63:48)
  g1[2] = (int)(((Kdim >> 16) & 0xFFFFu) | ((rows_total & 0xFFFFu) << 16)); // dim0 hi / dim1 lo
  g1[3] = (int)(((rows_total >> 16) & 0xFFFFu) | (32u << 16));       // dim1 hi / tile_dim0=32
  g1[4] = 128;                                                       // tile_dim1=128, tile_dim2=0
  g1[5] = (int)Kdim;                                                 // tensor_dim0_stride lo32 (elems)
  g1[6] = 0;                                                         // stride hi16 / dim1_stride lo16
  g1[7] = 0;
  const v4i z4 = {0, 0, 0, 0};
  const v8i z8 = {0, 0, 0, 0, 0, 0, 0, 0};
  __builtin_amdgcn_tensor_load_to_lds(g0, g1, z4, z4, z8, 0);
}
#endif

// ---------------------------------------------------------------------------
// f32 -> bf16 elementwise (weight conversion)
// ---------------------------------------------------------------------------
__global__ __launch_bounds__(256) void f32_to_bf16_k(const float* __restrict__ in,
                                                     uint16_t* __restrict__ out, int n) {
  int i = blockIdx.x * 256 + threadIdx.x;
  if (i < n) out[i] = to_bf16(in[i]);
}

// ---------------------------------------------------------------------------
// LayerNorm over C=2048, one row per workgroup, bf16 output
// ---------------------------------------------------------------------------
__global__ __launch_bounds__(256) void layernorm_bf16_k(const float* __restrict__ x,
                                                        const float* __restrict__ g,
                                                        const float* __restrict__ b,
                                                        uint16_t* __restrict__ o) {
  const int row = blockIdx.x;
  const int tid = threadIdx.x;
  const float* xr = x + (size_t)row * 2048;
  float v[8], s = 0.f, ss = 0.f;
#pragma unroll
  for (int i = 0; i < 8; ++i) {
    v[i] = xr[tid + i * 256];
    s += v[i]; ss += v[i] * v[i];
  }
  __shared__ float rs[256], rq[256];
  rs[tid] = s; rq[tid] = ss;
  __syncthreads();
  for (int st = 128; st > 0; st >>= 1) {
    if (tid < st) { rs[tid] += rs[tid + st]; rq[tid] += rq[tid + st]; }
    __syncthreads();
  }
  const float mu = rs[0] * (1.0f / 2048.0f);
  const float var = rq[0] * (1.0f / 2048.0f) - mu * mu;
  const float rstd = rsqrtf(var + 1e-5f);
  uint16_t* orow = o + (size_t)row * 2048;
#pragma unroll
  for (int i = 0; i < 8; ++i) {
    int c = tid + i * 256;
    orow[c] = to_bf16((v[i] - mu) * rstd * g[c] + b[c]);
  }
}

// ---------------------------------------------------------------------------
// Tiled WMMA GEMM:  out[M,N] = act( A[M,K] @ W[N,K]^T + bias ) (+ resid)
// 128x128 tile / workgroup, 8 waves in 2(M) x 4(N), each wave 64x32 (8 WMMA accs)
// Tile staging via Tensor Data Mover (async, TENSORcnt) when available.
// ACT: 0 = none, 1 = exact GELU.  OUTBF16: write bf16 else f32.
// ---------------------------------------------------------------------------
template <int ACT, bool RESID, bool OUTBF16>
__global__ __launch_bounds__(256) void gemm_bf16_k(const uint16_t* __restrict__ A,
                                                   const uint16_t* __restrict__ W,
                                                   const float* __restrict__ bias,
                                                   const float* __restrict__ resid,
                                                   float* __restrict__ outF,
                                                   uint16_t* __restrict__ outB,
                                                   int M, int N, int K) {
  __shared__ __align__(16) uint16_t As[2][128][40];  // 32 used cols + 8 pad
  __shared__ __align__(16) uint16_t Ws[2][128][40];

  const int tid  = threadIdx.x;
  const int wave = tid >> 5, lane = tid & 31;
  const int wm = wave >> 2;          // 0..1  -> 64 rows
  const int wn = wave & 3;           // 0..3  -> 32 cols
  const int hf = lane >> 4;
  const int n0 = blockIdx.x * 128, m0 = blockIdx.y * 128;

  AccF acc[4][2];
#pragma unroll
  for (int i = 0; i < 4; ++i)
#pragma unroll
    for (int j = 0; j < 2; ++j)
#pragma unroll
      for (int g = 0; g < 8; ++g) acc[i][j].f[g] = 0.f;

  auto stage_tiles = [&](int buf, int k0) {
#if USE_TDM
    if (wave == 0) {  // one wave issues both tile DMAs; EXEC-independent, TENSORcnt-tracked
      tdm_load_tile_128x32((uint32_t)(uintptr_t)&As[buf][0][0],
                           &A[(size_t)m0 * K + k0], (uint32_t)K, (uint32_t)M);
      tdm_load_tile_128x32((uint32_t)(uintptr_t)&Ws[buf][0][0],
                           &W[(size_t)n0 * K + k0], (uint32_t)K, (uint32_t)N);
    }
#else
#pragma unroll
    for (int i = 0; i < 2; ++i) {
      int l = tid + i * 256;            // 0..511
      int row = l >> 2;                 // 0..127
      int ck  = (l & 3) * 8;            // 0,8,16,24
      *reinterpret_cast<uint4*>(&As[buf][row][ck]) =
          *reinterpret_cast<const uint4*>(&A[(size_t)(m0 + row) * K + k0 + ck]);
      *reinterpret_cast<uint4*>(&Ws[buf][row][ck]) =
          *reinterpret_cast<const uint4*>(&W[(size_t)(n0 + row) * K + k0 + ck]);
    }
#endif
  };

  const int nk = K >> 5;
  stage_tiles(0, 0);
#if USE_TDM
  __builtin_amdgcn_s_wait_tensorcnt(0);
#endif
  __syncthreads();

  for (int kt = 0; kt < nk; ++kt) {
    const int buf = kt & 1;
    if (kt + 1 < nk) stage_tiles(buf ^ 1, (kt + 1) * 32);  // async prefetch overlaps WMMA
    if (kt + 2 < nk) {  // L2 prefetch of tile-after-next (global_prefetch_b8)
      int row = tid & 127;
      __builtin_prefetch(&A[(size_t)(m0 + row) * K + (kt + 2) * 32 + (tid >> 7) * 16], 0, 1);
      __builtin_prefetch(&W[(size_t)(n0 + row) * K + (kt + 2) * 32 + (tid >> 7) * 16], 0, 1);
    }

    FragBF af[4], bfr[2];
#pragma unroll
    for (int i = 0; i < 4; ++i) load_frag_a(&As[buf][wm * 64 + i * 16][0], 40, af[i]);
#pragma unroll
    for (int j = 0; j < 2; ++j) load_frag_b(&Ws[buf][wn * 32 + j * 16][0], 40, bfr[j]);
#pragma unroll
    for (int i = 0; i < 4; ++i)
#pragma unroll
      for (int j = 0; j < 2; ++j) acc[i][j].v = wmma_bf16(af[i], bfr[j], acc[i][j].v);

#if USE_TDM
    __builtin_amdgcn_s_wait_tensorcnt(0);  // next-tile DMA complete (trivial for waves 1..7)
#endif
    __syncthreads();
  }

  // Epilogue. C layout: vgpr g -> row g (lanes 0-15) / g+8 (lanes 16-31), col = lane&15.
#pragma unroll
  for (int i = 0; i < 4; ++i)
#pragma unroll
    for (int j = 0; j < 2; ++j) {
      const int nn = n0 + wn * 32 + j * 16 + (lane & 15);
      const float bv = bias[nn];
#pragma unroll
      for (int g = 0; g < 8; ++g) {
        const int mm = m0 + wm * 64 + i * 16 + g + hf * 8;
        float val = acc[i][j].f[g] + bv;
        if (ACT == 1) val = 0.5f * val * (1.0f + erff(val * 0.70710678118f));
        const size_t idx = (size_t)mm * N + nn;
        if (RESID) val += resid[idx];
        if (OUTBF16) outB[idx] = to_bf16(val); else outF[idx] = val;
      }
    }
}

// ---------------------------------------------------------------------------
// Flash attention (causal), bf16 WMMA.
// qkv: [B*T, 6144] (Q | K | V per head blocks of 128).  y: [B*T, 2048]
// grid.x = B*H (32), grid.y = T/128 (16). 8 waves x 16 q-rows each.
// ---------------------------------------------------------------------------
__global__ __launch_bounds__(256) void attn_fwd_k(const uint16_t* __restrict__ qkv,
                                                  uint16_t* __restrict__ y) {
  constexpr int T = 2048, H = 16, R3C = 6144, C = 2048;
  __shared__ __align__(16) uint16_t Ks[32][136];    // 32 kv x 128 d (padded)
  __shared__ __align__(16) uint16_t Vs[128][40];    // transposed: d x 32 kv (padded)
  __shared__ __align__(16) uint16_t Ps[8][16][40];  // per-wave P tile 16 x 32 (padded)

  const int tid  = threadIdx.x;
  const int wave = tid >> 5, lane = tid & 31;
  const int r = lane & 15, hf = lane >> 4;
  const int bh = blockIdx.x;
  const int b = bh / H, h = bh % H;
  const int qtile = blockIdx.y;
  const int q_lo = qtile * 128 + wave * 16;  // first q-row of this wave
  const float sscale = 0.088388347648318447f; // 1/sqrt(128)

  // Load Q fragments (A layout) straight from global.
  FragBF qf[4];
  {
    const uint16_t* qrow = qkv + ((size_t)(b * T + q_lo + r)) * R3C + h * 128;
#pragma unroll
    for (int kk = 0; kk < 4; ++kk) {
      const uint16_t* p = qrow + kk * 32 + hf * 8;
      uint4 lo = *reinterpret_cast<const uint4*>(p);
      uint4 hi = *reinterpret_cast<const uint4*>(p + 16);
      qf[kk].u[0] = lo.x; qf[kk].u[1] = lo.y; qf[kk].u[2] = lo.z; qf[kk].u[3] = lo.w;
      qf[kk].u[4] = hi.x; qf[kk].u[5] = hi.y; qf[kk].u[6] = hi.z; qf[kk].u[7] = hi.w;
    }
  }

  AccF O[8];
  float m_s[8], l_s[8];
#pragma unroll
  for (int j = 0; j < 8; ++j) {
#pragma unroll
    for (int g = 0; g < 8; ++g) O[j].f[g] = 0.f;
  }
#pragma unroll
  for (int g = 0; g < 8; ++g) { m_s[g] = -__builtin_inff(); l_s[g] = 0.f; }

  const int kv_end = (qtile + 1) * 128;
  for (int kv0 = 0; kv0 < kv_end; kv0 += 32) {
    __syncthreads();  // prior iteration finished reading Ks/Vs/Ps

    // Cooperative stage: K tile (row-major) and V tile (transposed).
#pragma unroll
    for (int i = 0; i < 2; ++i) {
      int l = tid + i * 256;            // 0..511
      int kr = l >> 4, ck = (l & 15) * 8;
      *reinterpret_cast<uint4*>(&Ks[kr][ck]) = *reinterpret_cast<const uint4*>(
          &qkv[((size_t)(b * T + kv0 + kr)) * R3C + 2048 + h * 128 + ck]);
    }
#pragma unroll
    for (int i = 0; i < 8; ++i) {
      int u = tid + i * 256;            // 0..2047
      int kr = u >> 6, d = (u & 63) * 2;
      uint32_t w = *reinterpret_cast<const uint32_t*>(
          &qkv[((size_t)(b * T + kv0 + kr)) * R3C + 4096 + h * 128 + d]);
      Vs[d][kr]     = (uint16_t)(w & 0xFFFFu);
      Vs[d + 1][kr] = (uint16_t)(w >> 16);
    }
    __syncthreads();

    // S = Q @ K^T  (two 16x16 tiles covering 32 kv)
    AccF S[2];
#pragma unroll
    for (int n = 0; n < 2; ++n) {
#pragma unroll
      for (int g = 0; g < 8; ++g) S[n].f[g] = 0.f;
#pragma unroll
      for (int kk = 0; kk < 4; ++kk) {
        FragBF kf;
        load_frag_b(&Ks[n * 16][kk * 32], 136, kf);
        S[n].v = wmma_bf16(qf[kk], kf, S[n].v);
      }
    }

    // Online softmax update + write P (bf16) to per-wave LDS in A layout rows.
    const int col0 = kv0 + (lane & 15);
#pragma unroll
    for (int g = 0; g < 8; ++g) {
      const int qr = q_lo + g + hf * 8;
      float s0 = S[0].f[g] * sscale;
      float s1 = S[1].f[g] * sscale;
      if (col0 > qr)      s0 = -__builtin_inff();
      if (col0 + 16 > qr) s1 = -__builtin_inff();
      float rm = fmaxf(s0, s1);
#pragma unroll
      for (int off = 8; off > 0; off >>= 1) rm = fmaxf(rm, __shfl_xor(rm, off, 16));
      const float mn  = fmaxf(m_s[g], rm);
      const float mn2 = fmaxf(mn, -1e30f);   // avoid inf-inf on fully-masked tiles
      const float corr = __expf(m_s[g] - mn2);
      const float p0 = __expf(s0 - mn2);
      const float p1 = __expf(s1 - mn2);
      float rsum = p0 + p1;
#pragma unroll
      for (int off = 8; off > 0; off >>= 1) rsum += __shfl_xor(rsum, off, 16);
      l_s[g] = l_s[g] * corr + rsum;
      m_s[g] = mn;
#pragma unroll
      for (int j = 0; j < 8; ++j) O[j].f[g] *= corr;
      Ps[wave][g + hf * 8][lane & 15]        = to_bf16(p0);
      Ps[wave][g + hf * 8][16 + (lane & 15)] = to_bf16(p1);
    }
    __syncthreads();

    // O += P @ V   (P: 16x32 A-operand, V: 32x16 B-operand per 16-wide d tile)
    FragBF pf;
    load_frag_a(&Ps[wave][0][0], 40, pf);
#pragma unroll
    for (int j = 0; j < 8; ++j) {
      FragBF vf;
      load_frag_b(&Vs[j * 16][0], 40, vf);
      O[j].v = wmma_bf16(pf, vf, O[j].v);
    }
  }

  // Normalize and write y (bf16, [B*T, C] with head-major cols == reference reshape)
  float inv[8];
#pragma unroll
  for (int g = 0; g < 8; ++g) inv[g] = 1.0f / l_s[g];
#pragma unroll
  for (int j = 0; j < 8; ++j)
#pragma unroll
    for (int g = 0; g < 8; ++g) {
      const int trow = q_lo + g + hf * 8;
      y[((size_t)(b * T + trow)) * C + h * 128 + j * 16 + (lane & 15)] =
          to_bf16(O[j].f[g] * inv[g]);
    }
}

// ---------------------------------------------------------------------------
// Launch: LN1 -> QKV gemm -> attention -> proj gemm(+x) -> LN2 -> fc gemm(GELU)
//         -> proj2 gemm(+x2) -> d_out
// ---------------------------------------------------------------------------
extern "C" void kernel_launch(void* const* d_in, const int* in_sizes, int n_in,
                              void* d_out, int out_size, void* d_ws, size_t ws_size,
                              hipStream_t stream) {
  (void)in_sizes; (void)n_in; (void)out_size; (void)ws_size;
  const float* x      = (const float*)d_in[0];
  const float* ln1_g  = (const float*)d_in[1];
  const float* ln1_b  = (const float*)d_in[2];
  const float* attn_w = (const float*)d_in[3];
  const float* attn_b = (const float*)d_in[4];
  const float* proj_w = (const float*)d_in[5];
  const float* proj_b = (const float*)d_in[6];
  const float* ln2_g  = (const float*)d_in[7];
  const float* ln2_b  = (const float*)d_in[8];
  const float* fc_w   = (const float*)d_in[9];
  const float* fc_b   = (const float*)d_in[10];
  const float* p2_w   = (const float*)d_in[11];
  const float* p2_b   = (const float*)d_in[12];

  constexpr int C = 2048, T = 2048, Bb = 2, M = Bb * T;

  uint8_t* ws = (uint8_t*)d_ws;
  size_t off = 0;
  auto carve = [&](size_t bytes) -> void* {
    void* p = ws + off;
    off += (bytes + 255) & ~(size_t)255;
    return p;
  };
  uint16_t* wqkv = (uint16_t*)carve((size_t)3 * C * C * 2);
  uint16_t* wprj = (uint16_t*)carve((size_t)C * C * 2);
  uint16_t* wfc  = (uint16_t*)carve((size_t)4 * C * C * 2);
  uint16_t* wp2  = (uint16_t*)carve((size_t)4 * C * C * 2);
  uint16_t* xn   = (uint16_t*)carve((size_t)M * C * 2);        // LN output (reused)
  float*    x2   = (float*)   carve((size_t)M * C * 4);        // attn residual out
  uint16_t* qkvb = (uint16_t*)carve((size_t)M * 3 * C * 2);    // qkv (bf16)
  uint16_t* yb   = (uint16_t*)carve((size_t)M * C * 2);        // attn out (bf16)
  uint16_t* h3   = qkvb;  // reuse qkvb+yb region (exactly M*4C*2 bytes) for MLP hidden

  auto cvt = [&](const float* s, uint16_t* d, int n) {
    f32_to_bf16_k<<<dim3((unsigned)((n + 255) / 256)), 256, 0, stream>>>(s, d, n);
  };
  cvt(attn_w, wqkv, 3 * C * C);
  cvt(proj_w, wprj, C * C);
  cvt(fc_w,   wfc,  4 * C * C);
  cvt(p2_w,   wp2,  4 * C * C);

  // x -> LN1 -> xn (bf16)
  layernorm_bf16_k<<<dim3(M), 256, 0, stream>>>(x, ln1_g, ln1_b, xn);

  // qkv = xn @ attn_w^T + attn_b   (bf16 out)
  gemm_bf16_k<0, false, true><<<dim3(3 * C / 128, M / 128), 256, 0, stream>>>(
      xn, wqkv, attn_b, nullptr, nullptr, qkvb, M, 3 * C, C);

  // y = attention(qkv)  (bf16 out)
  attn_fwd_k<<<dim3(Bb * 16, T / 128), 256, 0, stream>>>(qkvb, yb);

  // x2 = y @ proj_w^T + proj_b + x   (f32 out)
  gemm_bf16_k<0, true, false><<<dim3(C / 128, M / 128), 256, 0, stream>>>(
      yb, wprj, proj_b, x, x2, nullptr, M, C, C);

  // x2 -> LN2 -> xn (bf16)
  layernorm_bf16_k<<<dim3(M), 256, 0, stream>>>(x2, ln2_g, ln2_b, xn);

  // h3 = gelu(xn @ fc_w^T + fc_b)   (bf16 out)
  gemm_bf16_k<1, false, true><<<dim3(4 * C / 128, M / 128), 256, 0, stream>>>(
      xn, wfc, fc_b, nullptr, nullptr, h3, M, 4 * C, C);

  // out = h3 @ proj2_w^T + proj2_b + x2   (f32 out)
  gemm_bf16_k<0, true, false><<<dim3(C / 128, M / 128), 256, 0, stream>>>(
      h3, wp2, p2_b, x2, (float*)d_out, nullptr, M, C, 4 * C);
}
